// MultiHeadStateInpEncoder_14757507629767
// MI455X (gfx1250) — compile-verified, matrix operand order checked by reading, not run on hardware
//
#include <hip/hip_runtime.h>
#include <hip/hip_bf16.h>

// ---------------------------------------------------------------------------
// MultiHeadStateInpEncoder for MI455X (gfx1250): bf16 WMMA pipeline, v5
//  - all fragments from 16B contiguous chunks (global_load_b128 / ds_load_b128)
//  - weights + V pre-transposed so B fragments are K-contiguous
//  - fused single-pass QKV with TDM (tensor_load_to_lds, 6-arg builtin)
//    staging of the shared 32-row h strip into LDS (TENSORcnt + barrier)
//  - native __bf16 converts (v_cvt_pk_bf16_f32)
//  - global_prefetch_b8 on remaining streaming operands
// ---------------------------------------------------------------------------

typedef __attribute__((ext_vector_type(16))) __bf16 v16bf;
typedef __attribute__((ext_vector_type(8)))  __bf16 v8bf;
typedef __attribute__((ext_vector_type(8)))  float  v8f;
typedef __attribute__((ext_vector_type(4)))  unsigned u32x4;
typedef __attribute__((ext_vector_type(8)))  int      i32x8;
typedef __attribute__((ext_vector_type(4)))  int      i32x4;
typedef __bf16 bf16_t;

#define INDIM 80
#define HID   512
#define ATT   512
#define NH    8
#define HD    64
#define NA    32
#define NENT  64
#define NB    2048          // bs*t
#define ROWS1 (NB * NENT)   // 131072 rows into FC/QKV GEMMs
#define ROWSO (NB * NA)     // 65536 rows into output projection

// Workspace layout in bf16 elements.
// Weights stored TRANSPOSED: wT[n][k]  (wfcT is [512][96] with k>=80 zeroed)
static constexpr size_t OFF_WFC = 0;                       // 512*96
static constexpr size_t OFF_WQ  = OFF_WFC + 512 * 96;      // 512*512 each
static constexpr size_t OFF_WK  = OFF_WQ  + 512 * 512;
static constexpr size_t OFF_WV  = OFF_WK  + 512 * 512;
static constexpr size_t OFF_WO  = OFF_WV  + 512 * 512;
static constexpr size_t OFF_H   = OFF_WO  + 512 * 512;                 // 131072*512
static constexpr size_t OFF_Q   = OFF_H   + (size_t)ROWS1 * HID;       // [b,h,a,d]
static constexpr size_t OFF_K   = OFF_Q   + (size_t)NB * NH * NA * HD; // [b,h,e,d]
static constexpr size_t OFF_V   = OFF_K   + (size_t)NB * NH * NENT * HD; // [b,h,d,e] (transposed!)
static constexpr size_t OFF_O   = OFF_V   + (size_t)NB * NH * HD * NENT; // 65536*512
static constexpr size_t NWELEM  = 512 * 96 + 4 * 512 * 512;            // convert count

// Native convert (RNE on CDNA5); clang packs pairs into v_cvt_pk_bf16_f32
__device__ __forceinline__ __bf16 f2bf(float f) { return (__bf16)f; }

// Build one 16-bit A/B fragment from two contiguous 16B chunks.
// p must already include the +8*hi lane-half offset; chunks at K and K+16.
__device__ __forceinline__ v16bf load_frag(const bf16_t* p) {
  v8bf lo = *(const v8bf*)(p);
  v8bf hi = *(const v8bf*)(p + 16);
  return __builtin_shufflevector(lo, hi, 0, 1, 2, 3, 4, 5, 6, 7,
                                 8, 9, 10, 11, 12, 13, 14, 15);
}

__device__ __forceinline__ v8f wmma_bf16(v16bf a, v16bf b, v8f c) {
  return __builtin_amdgcn_wmma_f32_16x16x32_bf16(false, a, false, b, (short)0,
                                                 c, false, false);
}

// TDM: copy `elems` bf16 elements (1-D, contiguous) from global to LDS.
// D# built per CDNA5 ISA ch.8: group0 {count=1, lds_addr, global_addr, type=2},
// group1 {data_size=2B, tensor_dim0 = tile_dim0 = elems, tile_dim1 = 1}.
// This toolchain's builtin takes 6 args (g0, g1, g2, g3, extra group, cpol).
__device__ __forceinline__ void tdm_load_1d(const void* gsrc, void* lds_dst,
                                            unsigned elems) {
  unsigned long long ga = (unsigned long long)(uintptr_t)gsrc;
  unsigned lds = (unsigned)(uintptr_t)lds_dst;     // low 32 bits = LDS offset
  u32x4 g0 = { 1u,                                  // count=1, user mode
               lds,                                 // lds_addr [63:32]
               (unsigned)ga,                        // global_addr low
               (unsigned)((ga >> 32) & 0x01FFFFFFu) | (2u << 30) }; // type=2
  i32x8 g1 = { (int)0x00010000,                     // data_size=1 (2 bytes)
               (int)((elems & 0xFFFFu) << 16),      // tensor_dim0[15:0]
               (int)((elems >> 16) & 0xFFFFu) | (1 << 16), // dim0 hi | tensor_dim1=1
               (int)((elems & 0xFFFFu) << 16),      // tile_dim0
               1,                                   // tile_dim1 = 1
               (int)elems,                          // tensor_dim0_stride lo
               0, 0 };
  i32x4 gz4 = { 0, 0, 0, 0 };
  i32x8 gz8 = { 0, 0, 0, 0, 0, 0, 0, 0 };
  __builtin_amdgcn_tensor_load_to_lds(g0, g1, gz4, gz4, gz8, 0);
}

// ---------------------------------------------------------------------------
// Kernel 0: convert weights to bf16, TRANSPOSED (wT[n][k] = w[k][n]).
// wfcT is [512][96], rows k>=80 zero-padded.
// ---------------------------------------------------------------------------
__global__ void convert_weights_kernel(const float* __restrict__ wfc,
                                       const float* __restrict__ wq,
                                       const float* __restrict__ wk,
                                       const float* __restrict__ wv,
                                       const float* __restrict__ wo,
                                       bf16_t* __restrict__ ws) {
  size_t i = (size_t)blockIdx.x * blockDim.x + threadIdx.x;
  if (i >= NWELEM) return;
  if (i < 512 * 96) {
    int n = (int)(i / 96), k = (int)(i % 96);
    float v = (k < INDIM) ? wfc[(size_t)k * 512 + n] : 0.0f;
    ws[OFF_WFC + i] = f2bf(v);
  } else {
    size_t j   = i - 512 * 96;
    size_t sel = j >> 18;            // 262144-element segments
    size_t off = j & 262143u;
    int n = (int)(off >> 9), k = (int)(off & 511);
    const float* src = (sel == 0) ? wq : (sel == 1) ? wk : (sel == 2) ? wv : wo;
    ws[OFF_WQ + j] = f2bf(src[(size_t)k * 512 + n]);
  }
}

// ---------------------------------------------------------------------------
// Kernel 1: h = relu(x @ w_fc + b_fc)   (131072x80)x(80x512) -> bf16
// 32x64 tile per wave (2 m-tiles share each weight fragment). K padded to 96.
// ---------------------------------------------------------------------------
__global__ void fc_kernel(const float* __restrict__ x,
                          const bf16_t* __restrict__ wfcT,  // [512][96] bf16
                          const float* __restrict__ bfc,
                          bf16_t* __restrict__ h) {
  const int lane = threadIdx.x & 31;
  const int wave = (int)(((size_t)blockIdx.x * blockDim.x + threadIdx.x) >> 5);
  const int wp = wave >> 3, ns = wave & 7;    // wp in [0,4096): 32-row strip
  const int hi = lane >> 4, lm = lane & 15;
  const int row0 = wp * 32, n0 = ns * 64;

  const float* xr0 = x + (size_t)(row0 + lm) * INDIM + hi * 8;
  const float* xr1 = xr0 + (size_t)16 * INDIM;
  // prefetch both 320B input rows (lanes cover 32 distinct rows)
  for (int pb = 0; pb < 320; pb += 128) {
    __builtin_prefetch((const char*)xr0 + pb, 0, 3);
    __builtin_prefetch((const char*)xr1 + pb, 0, 3);
  }

  v8f acc[2][4] = {};
  const __bf16 z = f2bf(0.0f);
#pragma unroll
  for (int ks = 0; ks < 3; ++ks) {            // K chunks 0..95 (K>=80 zeroed)
    const int kb = ks * 32;
    v16bf a0, a1;
    {
      const v8f c0 = *(const v8f*)(xr0 + kb);
      const v8f d0 = *(const v8f*)(xr1 + kb);
#pragma unroll
      for (int e = 0; e < 8; ++e) { a0[e] = f2bf(c0[e]); a1[e] = f2bf(d0[e]); }
    }
    if (ks < 2) {                             // upper chunk valid for ks 0,1
      const v8f c1 = *(const v8f*)(xr0 + kb + 16);
      const v8f d1 = *(const v8f*)(xr1 + kb + 16);
#pragma unroll
      for (int e = 0; e < 8; ++e) { a0[8+e] = f2bf(c1[e]); a1[8+e] = f2bf(d1[e]); }
    } else {                                  // K 80..95: zero (don't read x)
#pragma unroll
      for (int e = 0; e < 8; ++e) { a0[8+e] = z; a1[8+e] = z; }
    }
#pragma unroll
    for (int nt = 0; nt < 4; ++nt) {
      const int n = n0 + nt * 16 + lm;
      v16bf bf = load_frag(wfcT + (size_t)n * 96 + kb + hi * 8);
      acc[0][nt] = wmma_bf16(a0, bf, acc[0][nt]);
      acc[1][nt] = wmma_bf16(a1, bf, acc[1][nt]);
    }
  }
#pragma unroll
  for (int mi = 0; mi < 2; ++mi) {
#pragma unroll
    for (int nt = 0; nt < 4; ++nt) {
      const int n = n0 + nt * 16 + lm;
      const float bias = bfc[n];
#pragma unroll
      for (int r = 0; r < 8; ++r) {
        const int m = row0 + mi * 16 + r + hi * 8;
        float v = acc[mi][nt][r] + bias;
        v = v > 0.0f ? v : 0.0f;
        h[(size_t)m * HID + n] = f2bf(v);
      }
    }
  }
}

// ---------------------------------------------------------------------------
// Kernel 2: fused QKV.  q = h@Wq+bq -> [b,h,a,d] (agents only)
//                       k = h@Wk+bk -> [b,h,e,d]
//                       v = relu(h@Wv+bv) -> TRANSPOSED [b,h,d,e]
// All 8 waves of a block share the same 32 h-rows (row0 = blockIdx*32):
// TDM-stage them into LDS once, A fragments come from ds_load_b128.
// Each A fragment feeds 12 WMMAs (3 matrices x 4 n-tiles).
// ---------------------------------------------------------------------------
__global__ void qkv_fused_kernel(const bf16_t* __restrict__ h,
                                 const bf16_t* __restrict__ wqT,
                                 const bf16_t* __restrict__ wkT,
                                 const bf16_t* __restrict__ wvT,
                                 const float* __restrict__ bq,
                                 const float* __restrict__ bk,
                                 const float* __restrict__ bv,
                                 bf16_t* __restrict__ dq,
                                 bf16_t* __restrict__ dk,
                                 bf16_t* __restrict__ dv) {
  __shared__ __align__(16) bf16_t sH[32 * HID];     // 32 rows x 1KB = 32KB
  const int lane = threadIdx.x & 31;
  const int wid  = threadIdx.x >> 5;                // wave in block: ns
  const int hi = lane >> 4, lm = lane & 15;
  const int row0 = blockIdx.x * 32;                 // shared by all 8 waves
  const int n0 = wid * 64;

  if (wid == 0) {                                   // one DMA per workgroup
    tdm_load_1d(h + (size_t)row0 * HID, &sH[0], 32 * HID);
    __builtin_amdgcn_s_wait_tensorcnt((short)0);
  }
  __syncthreads();

  const bf16_t* hr0 = &sH[0] + (size_t)lm * HID + hi * 8;
  const bf16_t* hr1 = hr0 + (size_t)16 * HID;

  v8f aq[2][4] = {}, ak[2][4] = {}, av[2][4] = {};
  for (int ks = 0; ks < 16; ++ks) {
    const int kb = ks * 32;
    v16bf a0 = load_frag(hr0 + kb);                 // ds_load_b128 x2
    v16bf a1 = load_frag(hr1 + kb);
#pragma unroll
    for (int nt = 0; nt < 4; ++nt) {
      const size_t wo_ = (size_t)(n0 + nt * 16 + lm) * HID + kb + hi * 8;
      v16bf fq = load_frag(wqT + wo_);
      aq[0][nt] = wmma_bf16(a0, fq, aq[0][nt]);
      aq[1][nt] = wmma_bf16(a1, fq, aq[1][nt]);
      v16bf fk = load_frag(wkT + wo_);
      ak[0][nt] = wmma_bf16(a0, fk, ak[0][nt]);
      ak[1][nt] = wmma_bf16(a1, fk, ak[1][nt]);
      v16bf fv = load_frag(wvT + wo_);
      av[0][nt] = wmma_bf16(a0, fv, av[0][nt]);
      av[1][nt] = wmma_bf16(a1, fv, av[1][nt]);
    }
  }
#pragma unroll
  for (int mi = 0; mi < 2; ++mi) {
#pragma unroll
    for (int nt = 0; nt < 4; ++nt) {
      const int n = n0 + nt * 16 + lm;
      const int head = n >> 6, d = n & 63;
      const float bsq = bq[n], bsk = bk[n], bsv = bv[n];
#pragma unroll
      for (int r = 0; r < 8; ++r) {
        const int m = row0 + mi * 16 + r + hi * 8;
        const int b = m >> 6, ent = m & 63;
        const size_t bh = (size_t)b * NH + head;
        if (ent < NA)
          dq[(bh * NA + ent) * HD + d] = f2bf(aq[mi][nt][r] + bsq);
        dk[(bh * NENT + ent) * HD + d] = f2bf(ak[mi][nt][r] + bsk);
        float vv = av[mi][nt][r] + bsv;
        vv = vv > 0.0f ? vv : 0.0f;
        dv[(bh * HD + d) * NENT + ent] = f2bf(vv);   // V transposed [b,h,d,e]
      }
    }
  }
}

// ---------------------------------------------------------------------------
// Kernel 3: attention. One wave per (b, head). S=QK^T, masked softmax, O=PV.
// K is [e][d] (contiguous d) for QK^T; V is [d][e] (contiguous e) for PV.
// ---------------------------------------------------------------------------
__global__ void attn_kernel(const bf16_t* __restrict__ q,
                            const bf16_t* __restrict__ kmat,
                            const bf16_t* __restrict__ vT,
                            const int* __restrict__ masks,
                            bf16_t* __restrict__ o) {
  __shared__ __align__(16) float  sS[4][NA * NENT];   // 4 x 8KB
  __shared__ __align__(16) bf16_t sP[4][NA * NENT];   // 4 x 4KB
  __shared__ int sM[4][NENT];
  const int w = threadIdx.x >> 5;
  const int lane = threadIdx.x & 31;
  const int hi = lane >> 4, lm = lane & 15;
  const size_t job = (size_t)blockIdx.x * 4 + w;  // 16384 jobs
  const int b = (int)(job >> 3), head = (int)(job & 7);

  sM[w][lane]      = masks[(size_t)b * NENT + lane];
  sM[w][lane + 32] = masks[(size_t)b * NENT + lane + 32];

  const bf16_t* qb = q    + ((size_t)b * NH + head) * (NA * HD);
  const bf16_t* kb = kmat + ((size_t)b * NH + head) * (NENT * HD);
  const bf16_t* vb = vT   + ((size_t)b * NH + head) * (HD * NENT);

  // Q fragments: rows mt*16+lm, K(head-dim) chunks
  v16bf qa[2][2];
#pragma unroll
  for (int mt = 0; mt < 2; ++mt)
#pragma unroll
    for (int ks = 0; ks < 2; ++ks)
      qa[mt][ks] = load_frag(qb + (size_t)(mt * 16 + lm) * HD + ks * 32 + hi * 8);

  // S = Q K^T / 8 -> LDS ; B fragments shared across both m-tiles
#pragma unroll
  for (int nt = 0; nt < 4; ++nt) {
    v16bf bk[2];
#pragma unroll
    for (int ks = 0; ks < 2; ++ks)
      bk[ks] = load_frag(kb + (size_t)(nt * 16 + lm) * HD + ks * 32 + hi * 8);
#pragma unroll
    for (int mt = 0; mt < 2; ++mt) {
      v8f acc = {};
#pragma unroll
      for (int ks = 0; ks < 2; ++ks)
        acc = wmma_bf16(qa[mt][ks], bk[ks], acc);
#pragma unroll
      for (int r = 0; r < 8; ++r)
        sS[w][(mt * 16 + r + hi * 8) * NENT + nt * 16 + lm] = acc[r] * 0.125f;
    }
  }
  __syncthreads();

  // Row softmax: lane a handles agent row a (mask rule: am[a] | m[n] => -1e30)
  {
    const int a = lane;
    const int am = sM[w][a];
    float mx = -3.0e38f;
    for (int n = 0; n < NENT; ++n) {
      float s = sS[w][a * NENT + n];
      if (am != 0 || sM[w][n] != 0) s = -1.0e30f;
      sS[w][a * NENT + n] = s;
      mx = fmaxf(mx, s);
    }
    float sum = 0.0f;
    for (int n = 0; n < NENT; ++n) {
      float e = __expf(sS[w][a * NENT + n] - mx);
      sS[w][a * NENT + n] = e;
      sum += e;
    }
    const float inv = 1.0f / sum;
    unsigned* prow = (unsigned*)&sP[w][a * NENT];
    for (int n = 0; n < NENT; n += 2) {       // packed pair stores
      float p0 = sS[w][a * NENT + n] * inv;
      float p1 = sS[w][a * NENT + n + 1] * inv;
      if (am != 0 || sM[w][n] != 0) p0 = 0.0f;
      if (am != 0 || sM[w][n + 1] != 0) p1 = 0.0f;
      unsigned lo_ = (unsigned)__builtin_bit_cast(unsigned short, f2bf(p0));
      unsigned hi_ = (unsigned)__builtin_bit_cast(unsigned short, f2bf(p1));
      prow[n >> 1] = lo_ | (hi_ << 16);
    }
  }
  __syncthreads();

  // O = P V -> o[b*32+a][head*64+d] ; V fragments shared across both m-tiles
  v16bf pa[2][2];
#pragma unroll
  for (int mt = 0; mt < 2; ++mt)
#pragma unroll
    for (int ks = 0; ks < 2; ++ks)
      pa[mt][ks] = load_frag(&sP[w][(mt * 16 + lm) * NENT + ks * 32 + hi * 8]);
#pragma unroll
  for (int nt = 0; nt < 4; ++nt) {
    v16bf bv[2];
#pragma unroll
    for (int ks = 0; ks < 2; ++ks)
      bv[ks] = load_frag(vb + (size_t)(nt * 16 + lm) * NENT + ks * 32 + hi * 8);
#pragma unroll
    for (int mt = 0; mt < 2; ++mt) {
      v8f acc = {};
#pragma unroll
      for (int ks = 0; ks < 2; ++ks)
        acc = wmma_bf16(pa[mt][ks], bv[ks], acc);
#pragma unroll
      for (int r = 0; r < 8; ++r) {
        const int a = mt * 16 + r + hi * 8;
        o[((size_t)b * NA + a) * ATT + head * HD + nt * 16 + lm] = f2bf(acc[r]);
      }
    }
  }
}

// ---------------------------------------------------------------------------
// Kernel 4: out = (o @ w_o + b_o), zeroed where agent mask set. f32 output.
// 32x64 tile per wave, shared B fragments.
// ---------------------------------------------------------------------------
__global__ void proj_kernel(const bf16_t* __restrict__ o,
                            const bf16_t* __restrict__ wT,  // [512][512]
                            const float* __restrict__ bias,
                            const int* __restrict__ masks,
                            float* __restrict__ out) {
  const int lane = threadIdx.x & 31;
  const int wave = (int)(((size_t)blockIdx.x * blockDim.x + threadIdx.x) >> 5);
  const int wp = wave >> 3, ns = wave & 7;    // wp in [0,2048)
  const int hi = lane >> 4, lm = lane & 15;
  const int row0 = wp * 32, n0 = ns * 64;

  const bf16_t* or0 = o + (size_t)(row0 + lm) * ATT + hi * 8;
  const bf16_t* or1 = or0 + (size_t)16 * ATT;
  for (int pb = 0; pb < 1024; pb += 128) {
    __builtin_prefetch((const char*)or0 + pb, 0, 3);
    __builtin_prefetch((const char*)or1 + pb, 0, 3);
  }

  v8f acc[2][4] = {};
  for (int ks = 0; ks < 16; ++ks) {
    const int kb = ks * 32;
    v16bf a0 = load_frag(or0 + kb);
    v16bf a1 = load_frag(or1 + kb);
#pragma unroll
    for (int nt = 0; nt < 4; ++nt) {
      const int n = n0 + nt * 16 + lm;
      v16bf bf = load_frag(wT + (size_t)n * HID + kb + hi * 8);
      acc[0][nt] = wmma_bf16(a0, bf, acc[0][nt]);
      acc[1][nt] = wmma_bf16(a1, bf, acc[1][nt]);
    }
  }
#pragma unroll
  for (int mi = 0; mi < 2; ++mi) {
#pragma unroll
    for (int nt = 0; nt < 4; ++nt) {
      const int n = n0 + nt * 16 + lm;
      const float bs = bias[n];
#pragma unroll
      for (int r = 0; r < 8; ++r) {
        const int m = row0 + mi * 16 + r + hi * 8;   // m = b*32 + a
        const int b = m >> 5, a = m & 31;
        float v = acc[mi][nt][r] + bs;
        if (masks[(size_t)b * NENT + a] != 0) v = 0.0f;
        out[(size_t)m * HID + n] = v;
      }
    }
  }
}

// ---------------------------------------------------------------------------
extern "C" void kernel_launch(void* const* d_in, const int* in_sizes, int n_in,
                              void* d_out, int out_size, void* d_ws, size_t ws_size,
                              hipStream_t stream) {
  (void)in_sizes; (void)n_in; (void)out_size; (void)ws_size;
  const float* inputs = (const float*)d_in[0];
  const int*   masks  = (const int*)d_in[1];
  const float* w_fc = (const float*)d_in[2];
  const float* b_fc = (const float*)d_in[3];
  const float* w_q  = (const float*)d_in[4];
  const float* b_q  = (const float*)d_in[5];
  const float* w_k  = (const float*)d_in[6];
  const float* b_k  = (const float*)d_in[7];
  const float* w_v  = (const float*)d_in[8];
  const float* b_v  = (const float*)d_in[9];
  const float* w_o  = (const float*)d_in[10];
  const float* b_o  = (const float*)d_in[11];
  bf16_t* ws  = (bf16_t*)d_ws;
  float*  out = (float*)d_out;

  convert_weights_kernel<<<(int)((NWELEM + 255) / 256), 256, 0, stream>>>(
      w_fc, w_q, w_k, w_v, w_o, ws);
  // FC: 4096 strip-pairs * 8 n-strips = 32768 waves = 4096 blocks x 256
  fc_kernel<<<4096, 256, 0, stream>>>(inputs, ws + OFF_WFC, b_fc, ws + OFF_H);
  // Fused QKV: one block per 32-row strip (TDM-staged), 8 n-strip waves each
  qkv_fused_kernel<<<4096, 256, 0, stream>>>(
      ws + OFF_H, ws + OFF_WQ, ws + OFF_WK, ws + OFF_WV, b_q, b_k, b_v,
      ws + OFF_Q, ws + OFF_K, ws + OFF_V);
  // Attention: 2048*8 = 16384 (b,head) jobs, 4 waves per 128-thread block
  attn_kernel<<<4096, 128, 0, stream>>>(ws + OFF_Q, ws + OFF_K, ws + OFF_V, masks,
                                        ws + OFF_O);
  // Proj: 2048 strip-pairs * 8 n-strips = 16384 waves = 2048 blocks x 256
  proj_kernel<<<2048, 256, 0, stream>>>(ws + OFF_O, ws + OFF_WO, b_o, masks, out);
}
